// SourceGraphRegressionModel_53730040873202
// MI455X (gfx1250) — matrix operand order repaired and checked by reading, not compiled, not run on hardware
//
#include <hip/hip_runtime.h>
#include <hip/hip_bf16.h>
#include <math.h>

// ---------------- CDNA5 WMMA types ----------------
typedef __attribute__((ext_vector_type(16))) __bf16 v16bf;
typedef __attribute__((ext_vector_type(8)))  float  v8f;

union BFrag { v16bf v; uint4 q[2]; unsigned short us[16]; };
union FAcc  { v8f   v; float f[8]; };

#define HID 128
#define LEAKY 0.2f

__device__ __forceinline__ unsigned short f2bf(float x) {
  unsigned int u = __float_as_uint(x);
  return (unsigned short)((u + 0x7FFFu + ((u >> 16) & 1u)) >> 16);
}
__device__ __forceinline__ float sigmoidf(float x) { return 1.0f / (1.0f + __expf(-x)); }

// ---------------------------------------------------------------------------
// Pack a K=128 x ncols row-major f32 weight matrix into WMMA B-fragment order:
// out[ tile j ][ kchunk ][ lane ][ 16 bf16 ]  (lane: N = 16j+(l&15); K contiguous)
// ---------------------------------------------------------------------------
__global__ void pack_b_kernel(const float* __restrict__ W, unsigned short* __restrict__ out,
                              int ncols, int nmats) {
  long long per = (long long)ncols * 128;
  long long t = (long long)blockIdx.x * blockDim.x + threadIdx.x;
  if (t >= per * nmats) return;
  int m   = (int)(t / per);
  int idx = (int)(t - (long long)m * per);
  int i = idx & 15;
  int l = (idx >> 4) & 31;
  int k = (idx >> 9) & 3;
  int j = idx >> 11;
  int kk = 32 * k + ((l < 16) ? 0 : 16) + i;
  int n  = 16 * j + (l & 15);
  out[(long long)m * per + idx] = f2bf(W[(long long)m * per + (long long)kk * ncols + n]);
}

// ---------------------------------------------------------------------------
// Char conv (KW=5, SAME) + relu + max over positions -> label_rep (U x 64)
// ---------------------------------------------------------------------------
__global__ void conv_label_kernel(const int* __restrict__ labels, const float* __restrict__ char_embed,
                                  const float* __restrict__ conv_w, const float* __restrict__ conv_b,
                                  float* __restrict__ label_rep, int U) {
  __shared__ float w_s[5 * 16 * 64];
  __shared__ float ch_s[16 * 16];
  int tid = threadIdx.x;                       // 0..63 = output channel
  for (int i = tid; i < 5 * 16 * 64; i += 64) w_s[i] = conv_w[i];
  float bc = conv_b[tid];
  int u0 = blockIdx.x * 16;
  for (int s = 0; s < 16; ++s) {
    int u = u0 + s;
    if (u >= U) break;
    __syncthreads();
    for (int i = tid; i < 16 * 16; i += 64) {
      int p = i >> 4, ci = i & 15;
      int code = labels[u * 16 + p] & 255;
      ch_s[i] = char_embed[code * 16 + ci];
    }
    __syncthreads();
    float mx = -1e30f;
    for (int p = 0; p < 16; ++p) {
      float acc = bc;
      #pragma unroll
      for (int kw = 0; kw < 5; ++kw) {
        int pp = p + kw - 2;
        if (pp < 0 || pp >= 16) continue;
        const float* cw = &w_s[(kw * 16) * 64 + tid];
        const float* cs = &ch_s[pp * 16];
        #pragma unroll
        for (int ci = 0; ci < 16; ++ci) acc = fmaf(cs[ci], cw[ci * 64], acc);
      }
      mx = fmaxf(mx, acc);
    }
    label_rep[u * 64 + tid] = fmaxf(mx, 0.0f);
  }
}

// h[n, :64] = label_rep[idx[n]]; h[n, 64:] = 0; also bf16 copy for msg gathers
__global__ void init_h_kernel(const float* __restrict__ label_rep, const int* __restrict__ idx,
                              float* __restrict__ h, unsigned short* __restrict__ hbf, int N) {
  long long t = (long long)blockIdx.x * blockDim.x + threadIdx.x;
  if (t >= (long long)N * 128) return;
  int n = (int)(t >> 7), c = (int)(t & 127);
  float v = (c < 64) ? label_rep[(long long)idx[n] * 64 + c] : 0.0f;
  h[t] = v;
  hbf[t] = f2bf(v);
}

// ---------------------------------------------------------------------------
// Load bf16 A fragment for rows of h_bf16 (lane = row l&15; K contiguous, k = 16*(l/16) + s)
// ---------------------------------------------------------------------------
__device__ __forceinline__ void load_a_bf16(BFrag a[4], const unsigned short* __restrict__ row, int ks) {
  #pragma unroll
  for (int k = 0; k < 4; ++k) {
    a[k].q[0] = *(const uint4*)(row + 32 * k + ks);
    a[k].q[1] = *(const uint4*)(row + 32 * k + ks + 8);
  }
}

// ---------------------------------------------------------------------------
// Message passing: per block = 256 edges of one etype; 8 waves x 32 edges
// (two 16-edge A-tiles per wave so every LDS B fragment feeds 2 WMMAs).
// agg[tgt] += h_bf16[src] @ W  via v_wmma_f32_16x16x32_bf16 + atomic f32.
// Scatter pointers precomputed once; per-tile address = ptr + 64*j immediate.
// ---------------------------------------------------------------------------
struct AdjArg { const int* adj[12]; int ecount[12]; };

__global__ __launch_bounds__(256) void msg_kernel(const unsigned short* __restrict__ hbf,
                                                  const unsigned short* __restrict__ Wp_layer,
                                                  AdjArg aa, float* __restrict__ agg) {
  int t = blockIdx.y;
  int E = aa.ecount[t];
  int e0 = blockIdx.x * 256;
  if (e0 >= E) return;
  extern __shared__ unsigned short wlds[];               // 16384 ushorts = 32KB
  {
    const uint4* s4 = (const uint4*)(Wp_layer + (long long)t * (128 * 128));
    uint4* d4 = (uint4*)wlds;
    for (int i = threadIdx.x; i < 2048; i += 256) d4[i] = s4[i];
  }
  __syncthreads();

  int lane = threadIdx.x & 31;
  int wave = threadIdx.x >> 5;
  int eb0 = e0 + wave * 32;                              // tile0 edges
  if (eb0 >= E) return;                                  // idle wave (no more barriers)
  int eb1 = eb0 + 16;                                    // tile1 edges
  int lr  = lane & 15;
  int ks  = (lane < 16) ? 0 : 16;
  int half = (lane < 16) ? 0 : 8;
  const int* adj = aa.adj[t];

  int i0 = eb0 + lr; if (i0 >= E) i0 = E - 1;
  int i1 = eb1 + lr; if (i1 >= E) i1 = E - 1;
  int src0 = adj[i0 * 2 + 0], tgt0 = adj[i0 * 2 + 1];
  int src1 = adj[i1 * 2 + 0], tgt1 = adj[i1 * 2 + 1];

  BFrag a0[4], a1[4];
  load_a_bf16(a0, hbf + (long long)src0 * 128, ks);
  load_a_bf16(a1, hbf + (long long)src1 * 128, ks);

  // Precompute full scatter pointers (col base = lane&15); tile j adds 64*j bytes.
  float *p0[8], *p1[8];
  #pragma unroll
  for (int r = 0; r < 8; ++r) {
    p0[r] = agg + (long long)__shfl(tgt0, r + half, 32) * 128 + lr;
    p1[r] = agg + (long long)__shfl(tgt1, r + half, 32) * 128 + lr;
  }

  if (eb1 + 16 <= E) {                                   // both tiles full (common)
    #pragma unroll
    for (int j = 0; j < 8; ++j) {
      FAcc c0, c1;
      #pragma unroll
      for (int r = 0; r < 8; ++r) { c0.f[r] = 0.0f; c1.f[r] = 0.0f; }
      #pragma unroll
      for (int k = 0; k < 4; ++k) {
        BFrag b;
        const uint4* bl = (const uint4*)(wlds + ((j * 4 + k) * 32 + lane) * 16);
        b.q[0] = bl[0]; b.q[1] = bl[1];
        c0.v = __builtin_amdgcn_wmma_f32_16x16x32_bf16(false, a0[k].v, false, b.v, (short)0, c0.v, false, false);
        c1.v = __builtin_amdgcn_wmma_f32_16x16x32_bf16(false, a1[k].v, false, b.v, (short)0, c1.v, false, false);
      }
      #pragma unroll
      for (int r = 0; r < 8; ++r) {
        __hip_atomic_fetch_add(p0[r] + 16 * j, c0.f[r], __ATOMIC_RELAXED, __HIP_MEMORY_SCOPE_AGENT);
        __hip_atomic_fetch_add(p1[r] + 16 * j, c1.f[r], __ATOMIC_RELAXED, __HIP_MEMORY_SCOPE_AGENT);
      }
    }
  } else {                                               // tail: guard each atomic
    #pragma unroll
    for (int j = 0; j < 8; ++j) {
      FAcc c0, c1;
      #pragma unroll
      for (int r = 0; r < 8; ++r) { c0.f[r] = 0.0f; c1.f[r] = 0.0f; }
      #pragma unroll
      for (int k = 0; k < 4; ++k) {
        BFrag b;
        const uint4* bl = (const uint4*)(wlds + ((j * 4 + k) * 32 + lane) * 16);
        b.q[0] = bl[0]; b.q[1] = bl[1];
        c0.v = __builtin_amdgcn_wmma_f32_16x16x32_bf16(false, a0[k].v, false, b.v, (short)0, c0.v, false, false);
        c1.v = __builtin_amdgcn_wmma_f32_16x16x32_bf16(false, a1[k].v, false, b.v, (short)0, c1.v, false, false);
      }
      #pragma unroll
      for (int r = 0; r < 8; ++r) {
        if (eb0 + r + half < E)
          __hip_atomic_fetch_add(p0[r] + 16 * j, c0.f[r], __ATOMIC_RELAXED, __HIP_MEMORY_SCOPE_AGENT);
        if (eb1 + r + half < E)
          __hip_atomic_fetch_add(p1[r] + 16 * j, c1.f[r], __ATOMIC_RELAXED, __HIP_MEMORY_SCOPE_AGENT);
      }
    }
  }
}

// ---------------------------------------------------------------------------
// Generic GEMM: C[M x (16*ntiles)] = act( A[M x 128] @ Bpacked + bias )
// A f32 row-major (lda), converted to bf16 fragments on load.
// 8 waves x 32 rows (two A-tiles) per block = 256 rows; B staged in LDS.
// ---------------------------------------------------------------------------
__global__ __launch_bounds__(256) void gemm_k128_kernel(const float* __restrict__ A, int lda, int M,
                                                        const unsigned short* __restrict__ Bp, int ntiles,
                                                        const float* __restrict__ bias,
                                                        float* __restrict__ C, int ldc, int act) {
  extern __shared__ unsigned short blds[];
  {
    const uint4* s4 = (const uint4*)Bp;
    uint4* d4 = (uint4*)blds;
    int n4 = ntiles * 256;
    for (int i = threadIdx.x; i < n4; i += 256) d4[i] = s4[i];
  }
  __syncthreads();

  int lane = threadIdx.x & 31;
  int wave = threadIdx.x >> 5;
  int row0 = blockIdx.x * 256 + wave * 32;               // tile0 rows
  if (row0 >= M) return;                                 // idle wave
  int row1 = row0 + 16;                                  // tile1 rows
  int lr  = lane & 15;
  int ks  = (lane < 16) ? 0 : 16;
  int half = (lane < 16) ? 0 : 8;

  int ar0 = row0 + lr; if (ar0 >= M) ar0 = M - 1;
  int ar1 = row1 + lr; if (ar1 >= M) ar1 = M - 1;
  const float* Ar0 = A + (long long)ar0 * lda;
  const float* Ar1 = A + (long long)ar1 * lda;

  BFrag a0[4], a1[4];
  #pragma unroll
  for (int k = 0; k < 4; ++k) {
    #pragma unroll
    for (int tile = 0; tile < 2; ++tile) {
      const float* Ar = tile ? Ar1 : Ar0;
      BFrag* a = tile ? &a1[k] : &a0[k];
      float4 x0 = *(const float4*)(Ar + 32 * k + ks);
      float4 x1 = *(const float4*)(Ar + 32 * k + ks + 4);
      float4 y0 = *(const float4*)(Ar + 32 * k + ks + 8);
      float4 y1 = *(const float4*)(Ar + 32 * k + ks + 12);
      a->us[0] = f2bf(x0.x); a->us[1] = f2bf(x0.y); a->us[2]  = f2bf(x0.z); a->us[3]  = f2bf(x0.w);
      a->us[4] = f2bf(x1.x); a->us[5] = f2bf(x1.y); a->us[6]  = f2bf(x1.z); a->us[7]  = f2bf(x1.w);
      a->us[8] = f2bf(y0.x); a->us[9] = f2bf(y0.y); a->us[10] = f2bf(y0.z); a->us[11] = f2bf(y0.w);
      a->us[12]= f2bf(y1.x); a->us[13]= f2bf(y1.y); a->us[14] = f2bf(y1.z); a->us[15] = f2bf(y1.w);
    }
  }

  // Precomputed store pointers; tile j adds 64*j bytes (immediate offset).
  float *p0[8], *p1[8];
  #pragma unroll
  for (int r = 0; r < 8; ++r) {
    p0[r] = C + (long long)(row0 + half + r) * ldc + lr;
    p1[r] = C + (long long)(row1 + half + r) * ldc + lr;
  }

  if (row1 + 16 <= M) {                                  // all 32 rows valid
    for (int j = 0; j < ntiles; ++j) {
      float bv = bias ? bias[16 * j + lr] : 0.0f;
      FAcc c0, c1;
      #pragma unroll
      for (int r = 0; r < 8; ++r) { c0.f[r] = bv; c1.f[r] = bv; }
      #pragma unroll
      for (int k = 0; k < 4; ++k) {
        BFrag b;
        const uint4* bl = (const uint4*)(blds + ((j * 4 + k) * 32 + lane) * 16);
        b.q[0] = bl[0]; b.q[1] = bl[1];
        c0.v = __builtin_amdgcn_wmma_f32_16x16x32_bf16(false, a0[k].v, false, b.v, (short)0, c0.v, false, false);
        c1.v = __builtin_amdgcn_wmma_f32_16x16x32_bf16(false, a1[k].v, false, b.v, (short)0, c1.v, false, false);
      }
      if (act == 1) {
        #pragma unroll
        for (int r = 0; r < 8; ++r) {
          c0.f[r] = (c0.f[r] > 0.0f) ? c0.f[r] : LEAKY * c0.f[r];
          c1.f[r] = (c1.f[r] > 0.0f) ? c1.f[r] : LEAKY * c1.f[r];
        }
      }
      #pragma unroll
      for (int r = 0; r < 8; ++r) {
        p0[r][16 * j] = c0.f[r];
        p1[r][16 * j] = c1.f[r];
      }
    }
  } else {                                               // tail rows: guard stores
    for (int j = 0; j < ntiles; ++j) {
      float bv = bias ? bias[16 * j + lr] : 0.0f;
      FAcc c0, c1;
      #pragma unroll
      for (int r = 0; r < 8; ++r) { c0.f[r] = bv; c1.f[r] = bv; }
      #pragma unroll
      for (int k = 0; k < 4; ++k) {
        BFrag b;
        const uint4* bl = (const uint4*)(blds + ((j * 4 + k) * 32 + lane) * 16);
        b.q[0] = bl[0]; b.q[1] = bl[1];
        c0.v = __builtin_amdgcn_wmma_f32_16x16x32_bf16(false, a0[k].v, false, b.v, (short)0, c0.v, false, false);
        c1.v = __builtin_amdgcn_wmma_f32_16x16x32_bf16(false, a1[k].v, false, b.v, (short)0, c1.v, false, false);
      }
      if (act == 1) {
        #pragma unroll
        for (int r = 0; r < 8; ++r) {
          c0.f[r] = (c0.f[r] > 0.0f) ? c0.f[r] : LEAKY * c0.f[r];
          c1.f[r] = (c1.f[r] > 0.0f) ? c1.f[r] : LEAKY * c1.f[r];
        }
      }
      #pragma unroll
      for (int r = 0; r < 8; ++r) {
        if (row0 + half + r < M) p0[r][16 * j] = c0.f[r];
        if (row1 + half + r < M) p1[r][16 * j] = c1.f[r];
      }
    }
  }
}

// gates1: z = sig(G1a+G2a) -> G1[:, :128];  rh = sig(G1b+G2b)*h -> G2[:, :128]
__global__ void gru_gates1_kernel(float* __restrict__ G1, float* __restrict__ G2,
                                  const float* __restrict__ h, int N) {
  long long t = (long long)blockIdx.x * blockDim.x + threadIdx.x;
  if (t >= (long long)N * 128) return;
  long long row = t >> 7; int c = (int)(t & 127);
  float g1a = G1[row * 256 + c],      g1b = G1[row * 256 + 128 + c];
  float g2a = G2[row * 256 + c],      g2b = G2[row * 256 + 128 + c];
  float hv  = h[row * 128 + c];
  float z = sigmoidf(g1a + g2a);
  float r = sigmoidf(g1b + g2b);
  G1[row * 256 + c] = z;
  G2[row * 256 + c] = r * hv;
}

// gates2: hh = tanh(G1[:,128:] + G2[:,128:]); h = z*h + (1-z)*hh; refresh bf16
__global__ void gru_gates2_kernel(const float* __restrict__ G1, const float* __restrict__ G2,
                                  float* __restrict__ h, unsigned short* __restrict__ hbf, int N) {
  long long t = (long long)blockIdx.x * blockDim.x + threadIdx.x;
  if (t >= (long long)N * 128) return;
  long long row = t >> 7; int c = (int)(t & 127);
  float z  = G1[row * 256 + c];
  float hh = tanhf(G1[row * 256 + 128 + c] + G2[row * 256 + 128 + c]);
  float hv = h[row * 128 + c];
  float hn = z * hv + (1.0f - z) * hh;
  h[row * 128 + c] = hn;
  hbf[row * 128 + c] = f2bf(hn);
}

// Readout: score=sig(h@sw+sb); v=leaky(V2@w3+b3)@w4+b4; out[g]+=score*v (LDS reduce)
__global__ void final_kernel(const float* __restrict__ h, const float* __restrict__ V2,
                             const int* __restrict__ n2g,
                             const float* __restrict__ sw, const float* __restrict__ sb,
                             const float* __restrict__ w3, const float* __restrict__ b3,
                             const float* __restrict__ w4, const float* __restrict__ b4,
                             float* __restrict__ out, int N, int G) {
  extern __shared__ float gsum[];
  for (int i = threadIdx.x; i < G; i += blockDim.x) gsum[i] = 0.0f;
  __syncthreads();
  long long n = (long long)blockIdx.x * blockDim.x + threadIdx.x;
  if (n < N) {
    const float* hr = h + n * 128;
    float s = sb[0];
    #pragma unroll 8
    for (int i = 0; i < 128; ++i) s = fmaf(hr[i], sw[i], s);
    float sc = sigmoidf(s);
    const float* v2 = V2 + n * 32;
    float v3[8];
    #pragma unroll
    for (int j = 0; j < 8; ++j) v3[j] = b3[j];
    for (int i = 0; i < 32; ++i) {
      float x = v2[i];
      #pragma unroll
      for (int j = 0; j < 8; ++j) v3[j] = fmaf(x, w3[i * 8 + j], v3[j]);
    }
    float v4 = b4[0];
    #pragma unroll
    for (int j = 0; j < 8; ++j) {
      float u = v3[j]; u = (u > 0.0f) ? u : LEAKY * u;
      v4 = fmaf(u, w4[j], v4);
    }
    atomicAdd(&gsum[n2g[n]], sc * v4);
  }
  __syncthreads();
  for (int i = threadIdx.x; i < G; i += blockDim.x)
    __hip_atomic_fetch_add(&out[i], gsum[i], __ATOMIC_RELAXED, __HIP_MEMORY_SCOPE_AGENT);
}

// ---------------------------------------------------------------------------
extern "C" void kernel_launch(void* const* d_in, const int* in_sizes, int n_in,
                              void* d_out, int out_size, void* d_ws, size_t ws_size,
                              hipStream_t stream) {
  const int*   labels_u   = (const int*)d_in[1];
  const int*   labels_idx = (const int*)d_in[2];
  const int*   n2g        = (const int*)d_in[3];
  const float* char_embed = (const float*)d_in[17];
  const float* conv_w     = (const float*)d_in[18];
  const float* conv_b     = (const float*)d_in[19];
  const float* W_msg      = (const float*)d_in[20];
  const float* gru_k      = (const float*)d_in[21];
  const float* gru_rk     = (const float*)d_in[22];
  const float* gru_b      = (const float*)d_in[23];
  const float* score_w    = (const float*)d_in[24];
  const float* score_b    = (const float*)d_in[25];
  const float* mlp_w1     = (const float*)d_in[26];
  const float* mlp_b1     = (const float*)d_in[27];
  const float* mlp_w2     = (const float*)d_in[28];
  const float* mlp_b2     = (const float*)d_in[29];
  const float* mlp_w3     = (const float*)d_in[30];
  const float* mlp_b3     = (const float*)d_in[31];
  const float* mlp_w4     = (const float*)d_in[32];
  const float* mlp_b4     = (const float*)d_in[33];

  int N = in_sizes[2];
  int U = in_sizes[1] / 16;
  AdjArg aa; int maxE = 0;
  for (int t = 0; t < 12; ++t) {
    aa.adj[t] = (const int*)d_in[5 + t];
    aa.ecount[t] = in_sizes[5 + t] / 2;
    if (aa.ecount[t] > maxE) maxE = aa.ecount[t];
  }

  // ---- workspace carve (~345MB; L2-resident hot state) ----
  char* base = (char*)d_ws; size_t off = 0;
  auto carve = [&](size_t bytes) -> void* {
    void* p = base + off; off += (bytes + 255) & ~(size_t)255; return p;
  };
  float*          label_rep = (float*)carve((size_t)U * 64 * 4);
  float*          h   = (float*)carve((size_t)N * 128 * 4);
  unsigned short* hbf = (unsigned short*)carve((size_t)N * 128 * 2);
  float*          agg = (float*)carve((size_t)N * 128 * 4);
  float*          G1  = (float*)carve((size_t)N * 256 * 4);
  float*          G2  = (float*)carve((size_t)N * 256 * 4);
  unsigned short* Wp  = (unsigned short*)carve((size_t)72 * 16384 * 2);
  unsigned short* kp  = (unsigned short*)carve((size_t)6 * 49152 * 2);
  unsigned short* rkp = (unsigned short*)carve((size_t)6 * 49152 * 2);
  unsigned short* w1p = (unsigned short*)carve((size_t)16384 * 2);
  unsigned short* w2p = (unsigned short*)carve((size_t)4096 * 2);
  if (off > ws_size) return;  // workspace too small: nothing safe to do

  // ---- pack all weights to WMMA B-fragment bf16 layout ----
  { long long tot = 72LL * 16384; pack_b_kernel<<<(int)((tot + 255) / 256), 256, 0, stream>>>(W_msg, Wp, 128, 72); }
  { long long tot = 6LL * 49152;  pack_b_kernel<<<(int)((tot + 255) / 256), 256, 0, stream>>>(gru_k, kp, 384, 6); }
  { long long tot = 6LL * 49152;  pack_b_kernel<<<(int)((tot + 255) / 256), 256, 0, stream>>>(gru_rk, rkp, 384, 6); }
  pack_b_kernel<<<64, 256, 0, stream>>>(mlp_w1, w1p, 128, 1);
  pack_b_kernel<<<16, 256, 0, stream>>>(mlp_w2, w2p, 32, 1);

  // ---- node features ----
  conv_label_kernel<<<(U + 15) / 16, 64, 0, stream>>>(labels_u, char_embed, conv_w, conv_b, label_rep, U);
  { long long tot = (long long)N * 128;
    init_h_kernel<<<(int)((tot + 255) / 256), 256, 0, stream>>>(label_rep, labels_idx, h, hbf, N); }

  int gblocks = (N + 255) / 256;
  long long ew = (long long)N * 128;
  int eblocks = (int)((ew + 255) / 256);

  for (int layer = 0; layer < 6; ++layer) {
    hipMemsetAsync(agg, 0, (size_t)N * 128 * 4, stream);
    dim3 mg((maxE + 255) / 256, 12);
    msg_kernel<<<mg, 256, 32768, stream>>>(hbf, Wp + (size_t)layer * 12 * 16384, aa, agg);

    const unsigned short* kpl  = kp  + (size_t)layer * 49152;
    const unsigned short* rkpl = rkp + (size_t)layer * 49152;
    const float* gb = gru_b + layer * 384;

    // g = agg @ gru_k + b  (first 256 cols, two 128-col passes)
    gemm_k128_kernel<<<gblocks, 256, 32768, stream>>>(agg, 128, N, kpl,            8, gb,       G1,       256, 0);
    gemm_k128_kernel<<<gblocks, 256, 32768, stream>>>(agg, 128, N, kpl + 8 * 2048, 8, gb + 128, G1 + 128, 256, 0);
    // h @ gru_rk (first 256 cols)
    gemm_k128_kernel<<<gblocks, 256, 32768, stream>>>(h, 128, N, rkpl,             8, nullptr,  G2,       256, 0);
    gemm_k128_kernel<<<gblocks, 256, 32768, stream>>>(h, 128, N, rkpl + 8 * 2048,  8, nullptr,  G2 + 128, 256, 0);
    gru_gates1_kernel<<<eblocks, 256, 0, stream>>>(G1, G2, h, N);
    // g3 = agg @ k3 + b3   (into G1[:,128:])
    gemm_k128_kernel<<<gblocks, 256, 32768, stream>>>(agg, 128, N, kpl + 16 * 2048, 8, gb + 256, G1 + 128, 256, 0);
    // (r*h) @ rk3          (rh lives in G2[:, :128], lda=256; into G2[:,128:])
    gemm_k128_kernel<<<gblocks, 256, 32768, stream>>>(G2, 256, N, rkpl + 16 * 2048, 8, nullptr,  G2 + 128, 256, 0);
    gru_gates2_kernel<<<eblocks, 256, 0, stream>>>(G1, G2, h, hbf, N);
  }

  // ---- readout MLP (V1 reuses agg, V2 reuses G2) ----
  gemm_k128_kernel<<<gblocks, 256, 32768, stream>>>(h,   128, N, w1p, 8, mlp_b1, agg, 128, 1);
  gemm_k128_kernel<<<gblocks, 256, 8192,  stream>>>(agg, 128, N, w2p, 2, mlp_b2, G2,  32,  1);

  hipMemsetAsync(d_out, 0, (size_t)out_size * 4, stream);
  final_kernel<<<(N + 255) / 256, 256, (size_t)out_size * 4, stream>>>(
      h, G2, n2g, score_w, score_b, mlp_w3, mlp_b3, mlp_w4, mlp_b4,
      (float*)d_out, N, out_size);
}
